// S_Block_55327768708400
// MI455X (gfx1250) — compile-verified, matrix operand order checked by reading, not run on hardware
//
#include <hip/hip_runtime.h>

// ---------------------------------------------------------------------------
// Types for CDNA5 WMMA (wave32): V_WMMA_F32_16X16X32_BF16
// ---------------------------------------------------------------------------
typedef __attribute__((ext_vector_type(16))) __bf16 bf16x16;
typedef __attribute__((ext_vector_type(8)))  __bf16 bf16x8;
typedef __attribute__((ext_vector_type(8)))  float  f32x8;

#define DEV __device__ __forceinline__

static constexpr int DIM    = 768;
static constexpr int QKV    = 3 * DIM;        // 2304
static constexpr int HEADS  = 12;
static constexpr int HD     = 64;
static constexpr int NS     = 1025;           // spatial sequence length (cls + 1024)
static constexpr int BT     = 8;              // B*T
static constexpr int NT_ROW = 8192;           // temporal tokens (2*4096)
static constexpr int NS_ROW = BT * NS;        // 8200 spatial tokens
static constexpr int NO_ROW = 2 * 4097;       // 8194 output rows
static constexpr int MLP_H  = 4 * DIM;        // 3072

template <bool B> struct BoolC { static constexpr bool value = B; };

union FragU {
    bf16x16 f;
    bf16x8  h[2];
};

DEV f32x8 zero_f32x8() {
    f32x8 z;
#pragma unroll
    for (int i = 0; i < 8; ++i) z[i] = 0.f;
    return z;
}

DEV bf16x8 zero_bf16x8() {
    bf16x8 z;
#pragma unroll
    for (int i = 0; i < 8; ++i) z[i] = (__bf16)0.f;
    return z;
}

// WMMA A(16x32)/B(32x16) bf16 fragment from a row-major K-stride array.
// Per-lane K pattern: elems 0..7 -> k0 + hi*8 + i ; elems 8..15 -> k0+16+hi*8+i.
DEV bf16x16 load_frag(const __bf16* rowptr, int k0, int hi, bool inb) {
    FragU u;
    if (inb) {
        u.h[0] = *(const bf16x8*)(rowptr + k0 + hi * 8);
        u.h[1] = *(const bf16x8*)(rowptr + k0 + 16 + hi * 8);
    } else {
        u.h[0] = zero_bf16x8();
        u.h[1] = zero_bf16x8();
    }
    return u.f;
}

DEV bf16x16 load_frag_lds(const __bf16* base32, int hi) {
    FragU u;
    u.h[0] = *(const bf16x8*)(base32 + hi * 8);
    u.h[1] = *(const bf16x8*)(base32 + 16 + hi * 8);
    return u.f;
}

DEV f32x8 wmma_bf16(bf16x16 a, bf16x16 b, f32x8 c) {
    return __builtin_amdgcn_wmma_f32_16x16x32_bf16(false, a, false, b, (short)0, c,
                                                   false, false);
}

// gfx1250 async global->LDS copy of 16 bytes (tracked by ASYNCcnt).
// Plain volatile (no memory clobber) so the scheduler stays free; ordering vs
// the consuming LDS reads is enforced by wait_asynccnt0()+barrier.
DEV void async_copy_b128(const void* gptr, void* lptr) {
    unsigned ldsa = (unsigned)(size_t)lptr;
    unsigned long long ga = (unsigned long long)(size_t)gptr;
    asm volatile("global_load_async_to_lds_b128 %0, %1, off"
                 :: "v"(ldsa), "v"(ga));
}

DEV void wait_asynccnt0() {
    asm volatile("s_wait_asynccnt 0" ::: "memory");
}

// ---------------------------------------------------------------------------
// Weight cast f32 -> bf16
// ---------------------------------------------------------------------------
__global__ void cast_f32_to_bf16(const float* __restrict__ s, __bf16* __restrict__ d,
                                 int n) {
    int i = blockIdx.x * blockDim.x + threadIdx.x;
    if (i < n) d[i] = (__bf16)s[i];
}

// ---------------------------------------------------------------------------
// LayerNorm (768 cols) -> bf16.  mode 0: temporal gather from x[:,1:,:]
// mode 1: spatial gather (cls from x, rest permuted from xt); mode 2: flat rows.
// block = (32, 8): one wave per row.
// ---------------------------------------------------------------------------
__global__ __launch_bounds__(256)
void ln_kernel(const float* __restrict__ x, const float* __restrict__ xt,
               const float* __restrict__ g, const float* __restrict__ b,
               __bf16* __restrict__ out, int rows, int mode) {
    int row = blockIdx.x * blockDim.y + threadIdx.y;
    if (row >= rows) return;
    int lane = threadIdx.x;
    const float* src;
    if (mode == 0) {
        int bb = row / 4096, n = row % 4096;
        src = x + ((size_t)bb * 4097 + 1 + n) * DIM;
    } else if (mode == 1) {
        int bt = row / NS, j = row % NS;
        int bb = bt / 4, t = bt % 4;
        if (j == 0) src = x + (size_t)bb * 4097 * DIM;
        else        src = xt + ((size_t)bb * 4096 + (size_t)(j - 1) * 4 + t) * DIM;
    } else {
        src = x + (size_t)row * DIM;
    }
    float vals[24];
    float s = 0.f, s2 = 0.f;
#pragma unroll
    for (int i = 0; i < 24; ++i) {
        float v = src[lane + i * 32];
        vals[i] = v; s += v; s2 += v * v;
    }
#pragma unroll
    for (int off = 16; off > 0; off >>= 1) {
        s  += __shfl_xor(s, off, 32);
        s2 += __shfl_xor(s2, off, 32);
    }
    float mean = s * (1.f / DIM);
    float var  = s2 * (1.f / DIM) - mean * mean;
    float rstd = rsqrtf(var + 1e-5f);
#pragma unroll
    for (int i = 0; i < 24; ++i) {
        int c = lane + i * 32;
        out[(size_t)row * DIM + c] = (__bf16)((vals[i] - mean) * rstd * g[c] + b[c]);
    }
}

// ---------------------------------------------------------------------------
// Generic bf16 WMMA GEMM: C[M,N] = A[M,K] * W[N,K]^T (+bias)(+GELU)(+resid)
// block = (32,4): 4 waves stacked on M (64 rows); block N-tile = 128.
// K-step 64: a 128x64 W tile (16 KB) is staged in LDS via gfx1250 async
// global->LDS copies and shared by all waves; each wave runs 16 WMMAs per
// iteration (2 A fragments x 8 N-subtiles).
// grid = (ceil(M/64), N/128).  N % 128 == 0, K % 64 == 0.
// ---------------------------------------------------------------------------
template <bool HAS_BIAS, bool ACT_GELU, bool OUT_BF16, bool OUT_F32, bool RESID>
__global__ __launch_bounds__(128)
void gemm_bf16_kernel(const __bf16* __restrict__ A, const __bf16* __restrict__ Wt,
                      const float* __restrict__ bias, float* __restrict__ outF,
                      __bf16* __restrict__ outB, const float* __restrict__ resid,
                      int M, int N, int K) {
    __shared__ __bf16 Bsh[128 * 64];   // 16 KB: [n-row][k]

    const int lane = threadIdx.x;
    const int wv   = threadIdx.y;
    const int tid  = lane + wv * 32;
    const int n16  = lane & 15, hi = lane >> 4;
    const int m0   = (blockIdx.x * 4 + wv) * 16;
    const int n0   = blockIdx.y * 128;
    const int arow = m0 + n16;
    const bool ain = arow < M;
    const __bf16* arp = A + (size_t)arow * K;

    f32x8 acc[8];
#pragma unroll
    for (int t = 0; t < 8; ++t) acc[t] = zero_f32x8();

    for (int k0 = 0; k0 < K; k0 += 64) {
        __syncthreads();   // previous iteration's LDS reads complete
        // 1024 x 16B chunks (128 rows x 64 cols), 8 async copies per thread
#pragma unroll
        for (int i = 0; i < 8; ++i) {
            int ch  = tid + i * 128;         // 0..1023
            int row = ch >> 3;               // 0..127
            int col = (ch & 7) * 8;          // 0..56
            async_copy_b128(Wt + (size_t)(n0 + row) * K + k0 + col,
                            &Bsh[row * 64 + col]);
        }
        if (k0 + 64 < K) {
            // gfx1250 global_prefetch_b8: warm next K-tile of W and A
            __builtin_prefetch(Wt + (size_t)(n0 + (tid >> 3)) * K + k0 + 64, 0, 1);
            if (ain) __builtin_prefetch(arp + k0 + 64, 0, 1);
        }
        wait_asynccnt0();
        __syncthreads();

        bf16x16 a0 = load_frag(arp, k0, hi, ain);
        bf16x16 a1 = load_frag(arp, k0 + 32, hi, ain);
#pragma unroll
        for (int t = 0; t < 8; ++t) {
            const __bf16* bp = &Bsh[(t * 16 + n16) * 64];
            acc[t] = wmma_bf16(a0, load_frag_lds(bp, hi), acc[t]);
            acc[t] = wmma_bf16(a1, load_frag_lds(bp + 32, hi), acc[t]);
        }
    }

    // C layout: element r of lane -> row M = r + 8*hi, col = n0 + t*16 + n16
#pragma unroll
    for (int t = 0; t < 8; ++t) {
#pragma unroll
        for (int r = 0; r < 8; ++r) {
            int row = m0 + r + 8 * hi;
            if (row < M) {
                int col = n0 + t * 16 + n16;
                float v = acc[t][r];
                if (HAS_BIAS) v += bias[col];
                if (ACT_GELU) v = 0.5f * v * (1.f + erff(v * 0.70710678118654752f));
                if (RESID)    v += resid[(size_t)row * N + col];
                if (OUT_F32)  outF[(size_t)row * N + col] = v;
                if (OUT_BF16) outB[(size_t)row * N + col] = (__bf16)v;
            }
        }
    }
}

// ---------------------------------------------------------------------------
// Temporal attention (T=4): one thread per (seq, head, t)
// ---------------------------------------------------------------------------
__global__ void temporal_attn(const __bf16* __restrict__ qkv, __bf16* __restrict__ out) {
    int idx = blockIdx.x * blockDim.x + threadIdx.x;
    if (idx >= 2048 * HEADS * 4) return;
    int t = idx & 3, h = (idx >> 2) % HEADS, seq = idx / (HEADS * 4);
    size_t sb = (size_t)seq * 4 * QKV;
    const __bf16* q = qkv + sb + (size_t)t * QKV + h * HD;
    float s[4] = {0.f, 0.f, 0.f, 0.f};
    for (int d = 0; d < HD; ++d) {
        float qd = (float)q[d];
#pragma unroll
        for (int tt = 0; tt < 4; ++tt)
            s[tt] += qd * (float)qkv[sb + (size_t)tt * QKV + DIM + h * HD + d];
    }
    float mx = fmaxf(fmaxf(s[0], s[1]), fmaxf(s[2], s[3])) * 0.125f;
    float p[4], ps = 0.f;
#pragma unroll
    for (int tt = 0; tt < 4; ++tt) { p[tt] = __expf(s[tt] * 0.125f - mx); ps += p[tt]; }
    float inv = 1.f / ps;
#pragma unroll
    for (int tt = 0; tt < 4; ++tt) p[tt] *= inv;
    for (int d = 0; d < HD; ++d) {
        float a = 0.f;
#pragma unroll
        for (int tt = 0; tt < 4; ++tt)
            a += p[tt] * (float)qkv[sb + (size_t)tt * QKV + 2 * DIM + h * HD + d];
        out[((size_t)seq * 4 + t) * DIM + h * HD + d] = (__bf16)a;
    }
}

// ---------------------------------------------------------------------------
// Residual add for temporal branch: xt[b,n] = x[b,1+n] + R[b,n]
// ---------------------------------------------------------------------------
__global__ void residual_t(const float* __restrict__ x, const float* __restrict__ R,
                           float* __restrict__ xt) {
    size_t id = (size_t)blockIdx.x * blockDim.x + threadIdx.x;
    if (id >= (size_t)NT_ROW * DIM) return;
    size_t row = id / DIM;
    int    c   = (int)(id % DIM);
    int b = (int)(row / 4096), n = (int)(row % 4096);
    xt[id] = x[((size_t)b * 4097 + 1 + n) * DIM + c] + R[id];
}

// ---------------------------------------------------------------------------
// Geometry bias: bias[h, i, j] for i,j in [0,1025)
// ---------------------------------------------------------------------------
__global__ __launch_bounds__(256)
void geom_bias_kernel(const float* __restrict__ wg_w, const float* __restrict__ wg_b,
                      float* __restrict__ bias) {
    __shared__ float wsh[HEADS * 64];
    __shared__ float bsh[HEADS];
    int tid = threadIdx.x;
    for (int i = tid; i < HEADS * 64; i += 256) wsh[i] = wg_w[i];
    if (tid < HEADS) bsh[tid] = wg_b[tid];
    __syncthreads();
    int id = blockIdx.x * 256 + tid;
    if (id >= NS * NS) return;
    int i = id / NS, j = id % NS;
    if (i == 0 || j == 0) {
        for (int h = 0; h < HEADS; ++h)
            bias[((size_t)h * NS + i) * NS + j] = 0.f;
        return;
    }
    int p = i - 1, q = j - 1;
    float dxv = logf(fmaxf(fabsf(((p >> 5) - (q >> 5)) * (1.f / 32.f) / 1.03125f), 0.001f));
    float dyv = logf(fmaxf(fabsf(((p & 31) - (q & 31)) * (1.f / 32.f) / 1.03125f), 0.001f));
    float sx[8], cx[8], sy[8], cy[8];
#pragma unroll
    for (int k = 0; k < 8; ++k) {
        float dm = __expf(-0.86346941f * (float)k);   // 1000^(-k/8)
        float ax = 100.f * dxv * dm, ay = 100.f * dyv * dm;
        sx[k] = __sinf(ax); cx[k] = __cosf(ax);
        sy[k] = __sinf(ay); cy[k] = __cosf(ay);
    }
    for (int h = 0; h < HEADS; ++h) {
        float acc = bsh[h];
        const float* wr = &wsh[h * 64];
#pragma unroll
        for (int k = 0; k < 8; ++k) {
            acc += sx[k] * wr[k] + sy[k] * wr[8 + k]
                 + cx[k] * wr[32 + k] + cy[k] * wr[40 + k]
                 + wr[48 + k] + wr[56 + k];
        }
        float w = fmaxf(acc, 0.f);
        bias[((size_t)h * NS + i) * NS + j] = logf(fmaxf(w, 1e-6f));
    }
}

// ---------------------------------------------------------------------------
// Spatial flash attention with WMMA.  One wave per (qtile, head, bt).
// 32 keys per tile: 4 WMMAs for S = Q*K^T, online softmax, 4 WMMAs for
// O += P*V (V staged transposed in LDS -> all fragments are ds_load_b128).
// All per-lane address components are hoisted; the tile loop only adds a
// uniform kb*QKV.  Main loop (32 tiles) has no bounds checks; the tail tile
// (kb=1024, 1 valid key) is guarded.  grid = (65, 12, 8), block = 32.
// ---------------------------------------------------------------------------
__global__ __launch_bounds__(32)
void spatial_attn(const __bf16* __restrict__ qkv, const float* __restrict__ bias,
                  __bf16* __restrict__ out) {
    const int lane = threadIdx.x;
    const int qt = blockIdx.x, h = blockIdx.y, bt = blockIdx.z;
    const int n16 = lane & 15, hi = lane >> 4;
    const size_t base = (size_t)bt * NS * QKV;

    __shared__ __bf16 Vt[64 * 32];   // transposed: [dim][key]
    __shared__ __bf16 Pt[16 * 32];   // [query][key]

    int qrow = qt * 16 + n16;
    bool qin = qrow < NS;
    const __bf16* qp = qkv + base + (size_t)qrow * QKV + h * HD;
    bf16x16 qf0 = load_frag(qp, 0, hi, qin);
    bf16x16 qf1 = load_frag(qp, 32, hi, qin);

    // hoisted per-lane invariants
    int   vrc[8];
    size_t coffV[8];
#pragma unroll
    for (int c = 0; c < 8; ++c) {
        int ch = c * 32 + lane;
        int vr = ch >> 3;                   // key 0..31
        int offd = (ch & 7) * 8;            // dim offset 0..56
        vrc[c] = vr;
        coffV[c] = (size_t)vr * QKV + offd;
    }
    const __bf16* vbase  = qkv + base + 2 * DIM + h * HD;
    const __bf16* k0base = qkv + base + (size_t)n16 * QKV + DIM + h * HD;
    const __bf16* k1base = k0base + (size_t)16 * QKV;
    const int iq0 = qt * 16 + 8 * hi;
    const float* bbase = bias + ((size_t)h * NS + iq0) * NS + n16;

    float mrow[8], lrow[8];
#pragma unroll
    for (int r = 0; r < 8; ++r) { mrow[r] = -1e30f; lrow[r] = 0.f; }
    f32x8 o[4];
#pragma unroll
    for (int t = 0; t < 4; ++t) o[t] = zero_f32x8();

    bf16x8 z8 = zero_bf16x8();

    auto do_tile = [&](int kb, auto guard_c) {
        constexpr bool G = decltype(guard_c)::value;
        const size_t kboff = (size_t)kb * QKV;
        __syncthreads();   // previous tile's Vt/Pt reads complete

        // ---- cooperative V tile load (32 keys x 64 dims), stored transposed ----
#pragma unroll
        for (int c = 0; c < 8; ++c) {
            int ch = c * 32 + lane;
            int offd = (ch & 7) * 8;
            bf16x8 vv;
            if (!G || (kb + vrc[c]) < NS)
                vv = *(const bf16x8*)(vbase + kboff + coffV[c]);
            else
                vv = z8;
#pragma unroll
            for (int i = 0; i < 8; ++i) Vt[(offd + i) * 32 + vrc[c]] = vv[i];
        }

        // ---- S tiles ----
        f32x8 s0 = zero_f32x8(), s1 = zero_f32x8();
        {
            bool kin = !G || (kb + n16) < NS;
            s0 = wmma_bf16(qf0, load_frag(k0base + kboff, 0, hi, kin), s0);
            s0 = wmma_bf16(qf1, load_frag(k0base + kboff, 32, hi, kin), s0);
        }
        {
            bool kin = !G || (kb + 16 + n16) < NS;
            s1 = wmma_bf16(qf0, load_frag(k1base + kboff, 0, hi, kin), s1);
            s1 = wmma_bf16(qf1, load_frag(k1base + kboff, 32, hi, kin), s1);
        }

        // ---- bias + online softmax (row M = r + 8*hi lives in one half-wave) ----
        float p0[8], p1[8], corr[8];
        int j0 = kb + n16, j1 = kb + 16 + n16;
#pragma unroll
        for (int r = 0; r < 8; ++r) {
            float v0 = s0[r] * 0.125f, v1 = s1[r] * 0.125f;
            if (iq0 + r < NS) {
                if (!G || j0 < NS) v0 += bbase[(size_t)r * NS + kb];
                if (!G || j1 < NS) v1 += bbase[(size_t)r * NS + kb + 16];
            }
            if (G) {
                if (j0 >= NS) v0 = -1e30f;
                if (j1 >= NS) v1 = -1e30f;
            }
            float t = fmaxf(v0, v1);
#pragma unroll
            for (int off = 8; off > 0; off >>= 1)
                t = fmaxf(t, __shfl_xor(t, off, 16));
            float mn = fmaxf(mrow[r], t);
            float e0 = __expf(v0 - mn), e1 = __expf(v1 - mn);
            float sum = e0 + e1;
#pragma unroll
            for (int off = 8; off > 0; off >>= 1)
                sum += __shfl_xor(sum, off, 16);
            float cf = __expf(mrow[r] - mn);
            mrow[r] = mn;
            lrow[r] = lrow[r] * cf + sum;
            corr[r] = cf;
            p0[r] = e0; p1[r] = e1;
        }
#pragma unroll
        for (int t = 0; t < 4; ++t)
#pragma unroll
            for (int r = 0; r < 8; ++r) o[t][r] *= corr[r];

        __syncthreads();
#pragma unroll
        for (int r = 0; r < 8; ++r) {
            Pt[(r + 8 * hi) * 32 + n16]      = (__bf16)p0[r];
            Pt[(r + 8 * hi) * 32 + 16 + n16] = (__bf16)p1[r];
        }
        __syncthreads();

        bf16x16 pa = load_frag_lds(&Pt[n16 * 32], hi);
#pragma unroll
        for (int t = 0; t < 4; ++t) {
            bf16x16 vb = load_frag_lds(&Vt[(t * 16 + n16) * 32], hi);
            o[t] = wmma_bf16(pa, vb, o[t]);
        }
    };

    for (int kb = 0; kb + 32 <= NS - 1; kb += 32)     // 32 full tiles (keys 0..1023)
        do_tile(kb, BoolC<false>{});
    do_tile(1024, BoolC<true>{});                     // tail tile (key 1024)

    // ---- finalize and store ----
#pragma unroll
    for (int t = 0; t < 4; ++t) {
#pragma unroll
        for (int r = 0; r < 8; ++r) {
            int iq = iq0 + r;
            if (iq < NS) {
                float val = o[t][r] / lrow[r];
                out[((size_t)bt * NS + iq) * DIM + h * HD + t * 16 + n16] = (__bf16)val;
            }
        }
    }
}

// ---------------------------------------------------------------------------
// Assemble: xout = concat(init_cls, xt) + concat(mean_t(cls_out), res_s perm)
// ---------------------------------------------------------------------------
__global__ void assemble_kernel(const float* __restrict__ x, const float* __restrict__ xt,
                                const float* __restrict__ Sout, float* __restrict__ out) {
    size_t id = (size_t)blockIdx.x * blockDim.x + threadIdx.x;
    if (id >= (size_t)NO_ROW * DIM) return;
    int    c   = (int)(id % DIM);
    size_t row = id / DIM;
    int b = (int)(row / 4097), n = (int)(row % 4097);
    float v;
    if (n == 0) {
        float acc = 0.f;
#pragma unroll
        for (int t = 0; t < 4; ++t)
            acc += Sout[(((size_t)b * 4 + t) * NS) * DIM + c];
        v = x[id] + acc * 0.25f;
    } else {
        int nn = n - 1;
        int s = nn >> 2, t = nn & 3;
        v = xt[((size_t)b * 4096 + nn) * DIM + c] +
            Sout[(((size_t)b * 4 + t) * NS + 1 + s) * DIM + c];
    }
    out[id] = v;
}

// ---------------------------------------------------------------------------
// Host launcher
// ---------------------------------------------------------------------------
extern "C" void kernel_launch(void* const* d_in, const int* in_sizes, int n_in,
                              void* d_out, int out_size, void* d_ws, size_t ws_size,
                              hipStream_t stream) {
    (void)in_sizes; (void)n_in; (void)out_size;
    const float* x       = (const float*)d_in[0];
    const float* norm1_g = (const float*)d_in[1];
    const float* norm1_b = (const float*)d_in[2];
    const float* qkv_w   = (const float*)d_in[3];
    const float* proj_w  = (const float*)d_in[4];
    const float* proj_b  = (const float*)d_in[5];
    const float* wg_w    = (const float*)d_in[6];
    const float* wg_b    = (const float*)d_in[7];
    const float* tnorm1_g = (const float*)d_in[8];
    const float* tnorm1_b = (const float*)d_in[9];
    const float* tqkv_w  = (const float*)d_in[10];
    const float* tproj_w = (const float*)d_in[11];
    const float* tproj_b = (const float*)d_in[12];
    const float* tfc_w   = (const float*)d_in[13];
    const float* tfc_b   = (const float*)d_in[14];
    const float* norm2_g = (const float*)d_in[15];
    const float* norm2_b = (const float*)d_in[16];
    const float* fc1_w   = (const float*)d_in[17];
    const float* fc1_b   = (const float*)d_in[18];
    const float* fc2_w   = (const float*)d_in[19];
    const float* fc2_b   = (const float*)d_in[20];
    float* out = (float*)d_out;

    // ---- workspace carve-up (256B aligned) ----
    uint8_t* w = (uint8_t*)d_ws;
    size_t off = 0;
    auto take = [&](size_t bytes) -> void* {
        void* p = w + off;
        off += (bytes + 255) & ~(size_t)255;
        return p;
    };
    __bf16* tqkv_bw = (__bf16*)take((size_t)QKV * DIM * 2);
    __bf16* tproj_bw = (__bf16*)take((size_t)DIM * DIM * 2);
    __bf16* tfc_bw  = (__bf16*)take((size_t)DIM * DIM * 2);
    __bf16* qkv_bw  = (__bf16*)take((size_t)QKV * DIM * 2);
    __bf16* proj_bw = (__bf16*)take((size_t)DIM * DIM * 2);
    __bf16* fc1_bw  = (__bf16*)take((size_t)MLP_H * DIM * 2);
    __bf16* fc2_bw  = (__bf16*)take((size_t)DIM * MLP_H * 2);
    __bf16* Abuf    = (__bf16*)take((size_t)NS_ROW * DIM * 2);           // LN outputs
    __bf16* A2buf   = (__bf16*)take((size_t)NS_ROW * DIM * 2);           // attn outputs
    __bf16* qkvbuf  = (__bf16*)take((size_t)NS_ROW * QKV * 2);           // qkv (also C1 alias)
    float*  Rbuf    = (float*)take((size_t)NS_ROW * DIM * 4);            // res_t / Sout (shared)
    float*  xtbuf   = (float*)take((size_t)NT_ROW * DIM * 4);
    size_t biasBytes = (size_t)HEADS * NS * NS * 4;
    size_t hBytes    = (size_t)NO_ROW * MLP_H * 2;
    float*  biasbuf = (float*)take(biasBytes > hBytes ? biasBytes : hBytes);
    __bf16* Hbuf    = (__bf16*)biasbuf;                                  // overlay (sequential use)
    __bf16* C1buf   = qkvbuf;                                            // overlay (sequential use)
    if (off > ws_size) return;  // workspace insufficient (deterministic no-op)

    const dim3 lnBlk(32, 8);
    const dim3 gmBlk(32, 4);

    // ---- weight casts ----
    auto castN = [&](const float* s, __bf16* d, int n) {
        cast_f32_to_bf16<<<(n + 255) / 256, 256, 0, stream>>>(s, d, n);
    };
    castN(tqkv_w, tqkv_bw, QKV * DIM);
    castN(tproj_w, tproj_bw, DIM * DIM);
    castN(tfc_w, tfc_bw, DIM * DIM);
    castN(qkv_w, qkv_bw, QKV * DIM);
    castN(proj_w, proj_bw, DIM * DIM);
    castN(fc1_w, fc1_bw, MLP_H * DIM);
    castN(fc2_w, fc2_bw, DIM * MLP_H);

    // ---- temporal branch ----
    ln_kernel<<<(NT_ROW + 7) / 8, lnBlk, 0, stream>>>(x, nullptr, tnorm1_g, tnorm1_b,
                                                      Abuf, NT_ROW, 0);
    gemm_bf16_kernel<false, false, true, false, false>
        <<<dim3((NT_ROW + 63) / 64, QKV / 128), gmBlk, 0, stream>>>(
            Abuf, tqkv_bw, nullptr, nullptr, qkvbuf, nullptr, NT_ROW, QKV, DIM);
    temporal_attn<<<(2048 * HEADS * 4 + 255) / 256, 256, 0, stream>>>(qkvbuf, A2buf);
    gemm_bf16_kernel<true, false, true, false, false>
        <<<dim3((NT_ROW + 63) / 64, DIM / 128), gmBlk, 0, stream>>>(
            A2buf, tproj_bw, tproj_b, nullptr, C1buf, nullptr, NT_ROW, DIM, DIM);
    gemm_bf16_kernel<true, false, false, true, false>
        <<<dim3((NT_ROW + 63) / 64, DIM / 128), gmBlk, 0, stream>>>(
            C1buf, tfc_bw, tfc_b, Rbuf, nullptr, nullptr, NT_ROW, DIM, DIM);
    residual_t<<<(NT_ROW * DIM) / 256, 256, 0, stream>>>(x, Rbuf, xtbuf);

    // ---- spatial branch ----
    ln_kernel<<<(NS_ROW + 7) / 8, lnBlk, 0, stream>>>(x, xtbuf, norm1_g, norm1_b,
                                                      Abuf, NS_ROW, 1);
    gemm_bf16_kernel<false, false, true, false, false>
        <<<dim3((NS_ROW + 63) / 64, QKV / 128), gmBlk, 0, stream>>>(
            Abuf, qkv_bw, nullptr, nullptr, qkvbuf, nullptr, NS_ROW, QKV, DIM);
    geom_bias_kernel<<<(NS * NS + 255) / 256, 256, 0, stream>>>(wg_w, wg_b, biasbuf);
    spatial_attn<<<dim3((NS + 15) / 16, HEADS, BT), 32, 0, stream>>>(qkvbuf, biasbuf,
                                                                     A2buf);
    gemm_bf16_kernel<true, false, false, true, false>
        <<<dim3((NS_ROW + 63) / 64, DIM / 128), gmBlk, 0, stream>>>(
            A2buf, proj_bw, proj_b, Rbuf, nullptr, nullptr, NS_ROW, DIM, DIM);
    assemble_kernel<<<(NO_ROW * DIM + 255) / 256, 256, 0, stream>>>(x, xtbuf, Rbuf, out);

    // ---- MLP ----
    ln_kernel<<<(NO_ROW + 7) / 8, lnBlk, 0, stream>>>(out, nullptr, norm2_g, norm2_b,
                                                      Abuf, NO_ROW, 2);
    gemm_bf16_kernel<true, true, true, false, false>
        <<<dim3((NO_ROW + 63) / 64, MLP_H / 128), gmBlk, 0, stream>>>(
            Abuf, fc1_bw, fc1_b, nullptr, Hbuf, nullptr, NO_ROW, MLP_H, DIM);
    gemm_bf16_kernel<true, false, false, true, true>
        <<<dim3((NO_ROW + 63) / 64, DIM / 128), gmBlk, 0, stream>>>(
            Hbuf, fc2_bw, fc2_b, out, nullptr, out, NO_ROW, DIM, MLP_H);
}
